// MemorizedAttention_11570641895986
// MI455X (gfx1250) — matrix operand order repaired and checked by reading, not compiled, not run on hardware
//
#include <hip/hip_runtime.h>
#include <hip/hip_bf16.h>
#include <stdint.h>

// MemorizedAttention for MI455X (gfx1250), wave32 + WMMA bf16.
// B=2 H=12 S=2048 D=64, memory length 300 (padded to 320 -> 37 chunks of 64).

#define B_      2
#define H_      12
#define S_      2048
#define D_      64
#define M_      300
#define MPAD_   320
#define KV_TOT_ (S_ + M_)        // 2348
#define KV_PAD_ (S_ + MPAD_)     // 2368
#define NCHUNK_ (KV_PAD_ / 64)   // 37
#define SCALE_  0.03608439182435161f   // 1/sqrt(768), hard-coded in reference

typedef __attribute__((ext_vector_type(16))) __bf16 v16bf;
typedef __attribute__((ext_vector_type(8)))  float  v8f;

union BFrag {
  v16bf v;
  uint4 u[2];
  __bf16 e[16];
};

#define WMMA_BF16(a, b, c) \
  __builtin_amdgcn_wmma_f32_16x16x32_bf16(false, (a), false, (b), (short)0, (c), false, false)

// B-fragment (32x16, K-contiguous source): 16 contiguous bf16 starting at p.
static __device__ inline v16bf load_b16(const __bf16* p) {
  BFrag f;
  f.u[0] = ((const uint4*)p)[0];
  f.u[1] = ((const uint4*)p)[1];
  return f.v;
}

// A-fragment (16x32) from a bf16 row-major row pointer (this lane's M row).
// e in [0,16): K = kbase + half*16 + e
static __device__ inline v16bf load_a_row(const __bf16* row, int kbase, int half) {
  BFrag f;
  f.u[0] = *(const uint4*)(row + kbase + half * 16);
  f.u[1] = *(const uint4*)(row + kbase + half * 16 + 8);
  return f.v;
}

// Same A-fragment but converting from an f32 row on the fly.
static __device__ inline v16bf load_a_f32row(const float* row, int kbase, int half) {
  BFrag f;
  const float* p = row + kbase + half * 16;
  const float* q = row + kbase + half * 16 + 8;
#pragma unroll
  for (int i = 0; i < 8; ++i) {
    f.e[i]     = (__bf16)p[i];
    f.e[8 + i] = (__bf16)q[i];
  }
  return f.v;
}

// ---------------------------------------------------------------------------
// Projection: out[s][n] = (sum_k X[s][k]*W[k][n] + b[n]) * outScale.
// transposed==0 : store bf16 row-major [BH][S][64]   (Q with scale, K)
// transposed==1 : store bf16 transposed [BH][64][S]  (V -> contiguous P@V B-frags)
// ---------------------------------------------------------------------------
__global__ __launch_bounds__(128) void proj_kernel(
    const float* __restrict__ X, const float* __restrict__ W,
    const float* __restrict__ bias, __bf16* __restrict__ out,
    int transposed, float outScale) {
  __shared__ __align__(16) __bf16 Wt[64 * 64];   // Wt[n][k] = W[k][n] (bf16)
  int tid = threadIdx.x;
  for (int i = tid; i < 64 * 64; i += 128) {
    int kk = i >> 6, nn = i & 63;
    Wt[nn * 64 + kk] = (__bf16)W[i];
  }
  __syncthreads();

  int lane = tid & 31, wave = tid >> 5;
  int half = lane >> 4, ln = lane & 15;
  int bh = blockIdx.y;
  int r0 = blockIdx.x * 64 + wave * 16;

  const float* xrow = X + ((size_t)bh * S_ + (r0 + ln)) * D_;
  v16bf a0 = load_a_f32row(xrow, 0, half);
  v16bf a1 = load_a_f32row(xrow, 32, half);

#pragma unroll
  for (int nt = 0; nt < 4; ++nt) {
    v8f acc = {};
    v16bf b0 = load_b16(&Wt[(nt * 16 + ln) * 64 + half * 16]);
    v16bf b1 = load_b16(&Wt[(nt * 16 + ln) * 64 + 32 + half * 16]);
    acc = WMMA_BF16(a0, b0, acc);
    acc = WMMA_BF16(a1, b1, acc);
    float bn = bias[nt * 16 + ln];
    if (!transposed) {
      __bf16* op = out + ((size_t)bh * S_ + r0 + half * 8) * D_ + nt * 16 + ln;
#pragma unroll
      for (int j = 0; j < 8; ++j) op[(size_t)j * D_] = (__bf16)((acc[j] + bn) * outScale);
    } else {
      __align__(16) __bf16 tmp[8];
#pragma unroll
      for (int j = 0; j < 8; ++j) tmp[j] = (__bf16)((acc[j] + bn) * outScale);
      __bf16* op = out + ((size_t)bh * D_ + nt * 16 + ln) * S_ + r0 + half * 8;
      *(uint4*)op = *(const uint4*)tmp;   // 8 contiguous bf16 (transposed store)
    }
  }
}

// ---------------------------------------------------------------------------
// Convert shared memory K/V (300x64 f32) -> bf16, zero-padded to 320 rows.
// Kmem: [320][64] row-major.  Vmemt: [64][320] transposed.
// ---------------------------------------------------------------------------
__global__ void convmem_kernel(const float* __restrict__ mk, const float* __restrict__ mv,
                               __bf16* __restrict__ kout, __bf16* __restrict__ vout) {
  int idx = blockIdx.x * 256 + threadIdx.x;
  if (idx >= MPAD_ * D_) return;
  int r = idx >> 6, c = idx & 63;
  float fk = 0.f, fv = 0.f;
  if (r < M_) { fk = mk[r * D_ + c]; fv = mv[r * D_ + c]; }
  kout[idx] = (__bf16)fk;
  vout[c * MPAD_ + r] = (__bf16)fv;
}

// ---------------------------------------------------------------------------
// Flash attention: 4 waves / WG, each wave owns 16 q rows x D=64.
// Per 64-key chunk: 8 WMMAs (QK^T), online softmax with half-wave shfl max
// reduction, P -> LDS -> two A-fragments, 2 WMMAs (row-sums via P @ ones),
// 8 WMMAs (P@V). Scale 1/sqrt(768) pre-folded into Q.
// ---------------------------------------------------------------------------
__global__ __launch_bounds__(128) void attn_kernel(
    const __bf16* __restrict__ Qbf, const __bf16* __restrict__ Kbf,
    const __bf16* __restrict__ Vt, const __bf16* __restrict__ Kmem,
    const __bf16* __restrict__ Vmemt, float* __restrict__ out) {
  __shared__ __align__(16) __bf16 pbuf[4][16][64];

  int tid = threadIdx.x;
  int lane = tid & 31, wave = tid >> 5;
  int half = lane >> 4, ln = lane & 15;
  int bh = blockIdx.z * H_ + blockIdx.y;
  int q0 = blockIdx.x * 64 + wave * 16;

  const __bf16* qrow = Qbf + ((size_t)bh * S_ + q0 + ln) * D_;
  v16bf qa0 = load_a_row(qrow, 0, half);
  v16bf qa1 = load_a_row(qrow, 32, half);

  BFrag onesf;   // all-ones B fragment: rowsum(P) = P @ ones, lands in C layout
#pragma unroll
  for (int i = 0; i < 16; ++i) onesf.e[i] = (__bf16)1.0f;

  const __bf16* Kb = Kbf + (size_t)bh * S_ * D_;
  const __bf16* Vb = Vt + (size_t)bh * D_ * S_;

  float mrow[8], lrow[8];
#pragma unroll
  for (int j = 0; j < 8; ++j) { mrow[j] = -3.0e38f; lrow[j] = 0.f; }
  v8f acc[4] = {};

  for (int c = 0; c < NCHUNK_; ++c) {
    int kb = c * 64;
    const __bf16 *ksrc, *vcol;
    size_t vstride;
    if (kb < S_) {                       // projected K/V region (uniform branch)
      ksrc = Kb + (size_t)kb * D_;
      vcol = Vb + kb;
      vstride = S_;
    } else {                             // shared-memory region (bf16, padded)
      int mb = kb - S_;
      ksrc = Kmem + (size_t)mb * D_;
      vcol = Vmemt + mb;
      vstride = MPAD_;
    }

    // --- scores for 4 column sub-tiles of 16 keys each (D=64 contraction) ---
    v8f s[4];
#pragma unroll
    for (int nt = 0; nt < 4; ++nt) {
      const __bf16* kp = ksrc + (size_t)(nt * 16 + ln) * D_;
      v16bf blo = load_b16(kp + half * 16);        // d 0..31
      v16bf bhi = load_b16(kp + 32 + half * 16);   // d 32..63
      v8f t = {};
      t = WMMA_BF16(qa0, blo, t);
      t = WMMA_BF16(qa1, bhi, t);
      s[nt] = t;
    }

    // --- online softmax (max via half-wave shfl; sums via WMMA below) ---
    bool needmask = (kb + 64 > KV_TOT_);   // only the last chunk
    float rj[8];
#pragma unroll
    for (int j = 0; j < 8; ++j) {
      float t0 = s[0][j], t1 = s[1][j], t2 = s[2][j], t3 = s[3][j];
      if (needmask) {
        if (kb + ln      >= KV_TOT_) t0 = -3.0e38f;
        if (kb + 16 + ln >= KV_TOT_) t1 = -3.0e38f;
        if (kb + 32 + ln >= KV_TOT_) t2 = -3.0e38f;
        if (kb + 48 + ln >= KV_TOT_) t3 = -3.0e38f;
      }
      float cm = fmaxf(fmaxf(t0, t1), fmaxf(t2, t3));
      cm = fmaxf(cm, __shfl_xor(cm, 1, 32));
      cm = fmaxf(cm, __shfl_xor(cm, 2, 32));
      cm = fmaxf(cm, __shfl_xor(cm, 4, 32));
      cm = fmaxf(cm, __shfl_xor(cm, 8, 32));
      float nm = fmaxf(mrow[j], cm);
      float r  = __expf(mrow[j] - nm);
      rj[j] = r;
      mrow[j] = nm;
      float e0 = __expf(t0 - nm);
      float e1 = __expf(t1 - nm);
      float e2 = __expf(t2 - nm);
      float e3 = __expf(t3 - nm);
      acc[0][j] *= r; acc[1][j] *= r; acc[2][j] *= r; acc[3][j] *= r;
      int row = half * 8 + j;
      pbuf[wave][row][ln]      = (__bf16)e0;
      pbuf[wave][row][16 + ln] = (__bf16)e1;
      pbuf[wave][row][32 + ln] = (__bf16)e2;
      pbuf[wave][row][48 + ln] = (__bf16)e3;
    }
    // LDS is in-order per wave on CDNA5; wait for DS stores, fence compiler.
    asm volatile("s_wait_dscnt 0" ::: "memory");

    // C-layout -> two A-fragments (P is 16x64).
    BFrag pa0, pa1;
    pa0.u[0] = *(const uint4*)&pbuf[wave][ln][half * 16];
    pa0.u[1] = *(const uint4*)&pbuf[wave][ln][8 + half * 16];
    pa1.u[0] = *(const uint4*)&pbuf[wave][ln][32 + half * 16];
    pa1.u[1] = *(const uint4*)&pbuf[wave][ln][40 + half * 16];

    // Row sums of (bf16-rounded) P, replicated across lanes in C layout.
    v8f cs = {};
    cs = WMMA_BF16(pa0.v, onesf.v, cs);
    cs = WMMA_BF16(pa1.v, onesf.v, cs);
#pragma unroll
    for (int j = 0; j < 8; ++j) lrow[j] = lrow[j] * rj[j] + cs[j];

    // --- P @ V over 4 d-tiles ---
#pragma unroll
    for (int nt = 0; nt < 4; ++nt) {
      const __bf16* vp = vcol + (size_t)(nt * 16 + ln) * vstride;
      acc[nt] = WMMA_BF16(pa0.v, load_b16(vp + half * 16), acc[nt]);
      acc[nt] = WMMA_BF16(pa1.v, load_b16(vp + 32 + half * 16), acc[nt]);
    }
  }

  float* orow = out + ((size_t)bh * S_ + q0) * D_;
#pragma unroll
  for (int j = 0; j < 8; ++j) {
    float inv = 1.0f / lrow[j];
    size_t ro = (size_t)(half * 8 + j) * D_;
    orow[ro + ln]      = acc[0][j] * inv;
    orow[ro + 16 + ln] = acc[1][j] * inv;
    orow[ro + 32 + ln] = acc[2][j] * inv;
    orow[ro + 48 + ln] = acc[3][j] * inv;
  }
}

extern "C" void kernel_launch(void* const* d_in, const int* in_sizes, int n_in,
                              void* d_out, int out_size, void* d_ws, size_t ws_size,
                              hipStream_t stream) {
  (void)in_sizes; (void)n_in; (void)out_size; (void)ws_size;
  const float* q  = (const float*)d_in[0];
  const float* k  = (const float*)d_in[1];
  const float* v  = (const float*)d_in[2];
  const float* Wq = (const float*)d_in[3];
  const float* bq = (const float*)d_in[4];
  const float* Wk = (const float*)d_in[5];
  const float* bk = (const float*)d_in[6];
  const float* Wv = (const float*)d_in[7];
  const float* bv = (const float*)d_in[8];
  const float* mk = (const float*)d_in[9];
  const float* mv = (const float*)d_in[10];

  char* ws = (char*)d_ws;
  const size_t bytesQKV = (size_t)B_ * H_ * S_ * D_ * 2;  // 6,291,456 (bf16)
  __bf16* Qbf   = (__bf16*)(ws);
  __bf16* Kbf   = (__bf16*)(ws + bytesQKV);
  __bf16* Vt    = (__bf16*)(ws + 2 * bytesQKV);
  __bf16* Kmem  = (__bf16*)(ws + 3 * bytesQKV);                          // [320][64]
  __bf16* Vmemt = (__bf16*)(ws + 3 * bytesQKV + (size_t)MPAD_ * D_ * 2); // [64][320]

  convmem_kernel<<<(MPAD_ * D_ + 255) / 256, 256, 0, stream>>>(mk, mv, Kmem, Vmemt);
  proj_kernel<<<dim3(S_ / 64, B_ * H_), 128, 0, stream>>>(q, Wq, bq, Qbf, 0, SCALE_);
  proj_kernel<<<dim3(S_ / 64, B_ * H_), 128, 0, stream>>>(k, Wk, bk, Kbf, 0, 1.0f);
  proj_kernel<<<dim3(S_ / 64, B_ * H_), 128, 0, stream>>>(v, Wv, bv, Vt, 1, 1.0f);
  attn_kernel<<<dim3(S_ / 64, H_, B_), 128, 0, stream>>>(Qbf, Kbf, Vt, Kmem, Vmemt,
                                                         (float*)d_out);
}
